// GRU_77996606095995
// MI455X (gfx1250) — compile-verified, matrix operand order checked by reading, not run on hardware
//
#include <hip/hip_runtime.h>

// GRU: T=512, B=64, H=512, L=2 on gfx1250.
// bf16 WMMA (v_wmma_f32_16x16x32_bf16) + TDM tensor_load_to_lds + 320KB-class LDS.

typedef __attribute__((ext_vector_type(16))) __bf16 v16bf;
typedef __attribute__((ext_vector_type(8)))  float  v8f;
typedef __attribute__((ext_vector_type(4)))  unsigned v4u;
typedef __attribute__((ext_vector_type(8)))  int      v8i_;
typedef __attribute__((ext_vector_type(4)))  int      v4i_;

#define T_  512
#define B_  64
#define H_  512
#define L_  2
#define M_  (T_ * B_)          // 32768 rows for input-side GEMM
#define NBLK_REC 16            // persistent recurrent blocks (co-resident)
#define BH_ (B_ * H_)

struct alignas(16) U4 { unsigned x, y, z, w; };
struct alignas(16) F4 { float x, y, z, w; };

__device__ __forceinline__ float sigmoidf_(float x) {
  return 1.0f / (1.0f + __expf(-x));
}

// ---- WMMA fragment helpers ------------------------------------------------
// B-fragment (32x16 K x N, bf16): per lane 16 contiguous bf16 (32B) in the
// pre-swizzled weight image (lane = N%16 + 16*K-half).
__device__ __forceinline__ v16bf b_load(const __bf16* p16, size_t fragbase, int lane) {
  union { U4 q[2]; v16bf v; } u;
  const U4* p = (const U4*)(p16 + fragbase + (size_t)lane * 16);
  u.q[0] = p[0];
  u.q[1] = p[1];
  return u.v;
}

// A-fragment (16x32 M x K) from a row-major bf16 image (LDS or global):
// lane row M=lane&15; elems 0..7 = K k0..k0+7, 8..15 = K k0+16..k0+23,
// k0 = 32*kt + 8*(lane>=16).
__device__ __forceinline__ v16bf a_from_bf16(const __bf16* img, int row, int k0) {
  union { U4 q[2]; v16bf v; } u;
  u.q[0] = *(const U4*)(img + (size_t)row * H_ + k0);
  u.q[1] = *(const U4*)(img + (size_t)row * H_ + k0 + 16);
  return u.v;
}

// Same A-fragment gathered from global fp32 with on-the-fly cvt.
__device__ __forceinline__ v16bf a_from_f32(const float* X, int row, int k0) {
  const F4* p = (const F4*)(X + (size_t)row * H_ + k0);
  F4 f0 = p[0], f1 = p[1], f2 = p[4], f3 = p[5];
  union { __bf16 h[16]; v16bf v; } u;
  u.h[0]  = (__bf16)f0.x; u.h[1]  = (__bf16)f0.y; u.h[2]  = (__bf16)f0.z; u.h[3]  = (__bf16)f0.w;
  u.h[4]  = (__bf16)f1.x; u.h[5]  = (__bf16)f1.y; u.h[6]  = (__bf16)f1.z; u.h[7]  = (__bf16)f1.w;
  u.h[8]  = (__bf16)f2.x; u.h[9]  = (__bf16)f2.y; u.h[10] = (__bf16)f2.z; u.h[11] = (__bf16)f2.w;
  u.h[12] = (__bf16)f3.x; u.h[13] = (__bf16)f3.y; u.h[14] = (__bf16)f3.z; u.h[15] = (__bf16)f3.w;
  return u.v;
}

// Weight fragment base: WB[(layer*6+gate)][ntile(32)][ktile(16)][lane*16+e]
__device__ __forceinline__ size_t wb_frag(int l, int g, int nt, int kt) {
  return ((((size_t)(l * 6 + g) * 32 + nt) * 16 + kt) * 512);
}

__device__ __forceinline__ v8f wmma_bf16(v16bf a, v16bf b, v8f c) {
  return __builtin_amdgcn_wmma_f32_16x16x32_bf16(false, a, false, b, (short)0, c,
                                                 false, false);
}

// ---- Tensor Data Mover: build D# per ISA ch.8 and issue the DMA -----------
// 2-D tile: tile0 elements (2B each) x tile1 rows; row stride = stride0 elems.
// Toolchain has the 6-arg builtin: (v4u, v8i, v4i, v4i, v8i, i32 cpol).
__device__ __forceinline__ void tdm_load(unsigned lds_byte, unsigned long long gaddr,
                                         unsigned tile0, unsigned tile1,
                                         unsigned tdim0, unsigned tdim1,
                                         unsigned long long stride0) {
  v4u g0;
  g0[0] = 1u;                                           // count=1, user mode
  g0[1] = lds_byte;                                     // lds_addr
  g0[2] = (unsigned)(gaddr & 0xFFFFFFFFull);            // global_addr[31:0]
  g0[3] = (unsigned)((gaddr >> 32) & 0x01FFFFFFull)     // global_addr[56:32]
        | (2u << 30);                                   // type=2 ("image")
  v8i_ g1;
  g1[0] = (int)(1u << 16);                              // data_size=1 (2B), mask=0
  g1[1] = (int)((tdim0 & 0xFFFFu) << 16);               // tensor_dim0[15:0]
  g1[2] = (int)(((tdim0 >> 16) & 0xFFFFu) |             // tensor_dim0[31:16]
                ((tdim1 & 0xFFFFu) << 16));             // tensor_dim1[15:0]
  g1[3] = (int)(((tdim1 >> 16) & 0xFFFFu) |             // tensor_dim1[31:16]
                ((tile0 & 0xFFFFu) << 16));             // tile_dim0
  g1[4] = (int)(tile1 & 0xFFFFu);                       // tile_dim1 (tile_dim2=0)
  g1[5] = (int)(stride0 & 0xFFFFFFFFull);               // tensor_dim0_stride lo
  g1[6] = (int)((stride0 >> 32) & 0xFFFFull);           // stride0 hi (stride1=0)
  g1[7] = 0;
  v4i_ z4 = {0, 0, 0, 0};
  v8i_ z8 = {0, 0, 0, 0, 0, 0, 0, 0};
  __builtin_amdgcn_tensor_load_to_lds(g0, g1, z4, z4, z8, 0);
}

// ---- grid barrier for persistent recurrent kernel -------------------------
__device__ __forceinline__ void gbar(unsigned* counter, unsigned phase) {
  __syncthreads();
  if (threadIdx.x == 0) {
    __threadfence();
    atomicAdd(counter, 1u);
    volatile unsigned* vc = counter;
    unsigned target = phase * (unsigned)NBLK_REC;
    while (*vc < target) { __builtin_amdgcn_s_sleep(1); }
    __threadfence();
  }
  __syncthreads();
}

// ---------------------------------------------------------------------------
// Kernel 1: fp32 -> bf16 weight swizzle into WMMA B-fragment layout.
// Gate order g: 0=ir 1=iz 2=in 3=hr 4=hz 5=hn. Source w[l][k][n] row-major.
// ---------------------------------------------------------------------------
__global__ void k_convert(const float* w_ir, const float* w_iz, const float* w_in,
                          const float* w_hr, const float* w_hz, const float* w_hn,
                          __bf16* WB, unsigned* counters) {
  size_t idx = (size_t)blockIdx.x * 256 + threadIdx.x;   // over L*6*H*H
  if (idx == 0) { counters[0] = 0u; counters[16] = 0u; } // reset grid barriers
  int lg = (int)(idx >> 18);          // / (H*H)
  int j  = (int)(idx & 262143);       // k*H + n
  int l = lg / 6, g = lg % 6;
  int k = j >> 9, n = j & 511;

  const float* src;
  switch (g) {
    case 0:  src = w_ir; break;
    case 1:  src = w_iz; break;
    case 2:  src = w_in; break;
    case 3:  src = w_hr; break;
    case 4:  src = w_hz; break;
    default: src = w_hn; break;
  }
  float val = src[(size_t)l * H_ * H_ + (size_t)j];

  int kk = k & 31, kt = k >> 5, nt = n >> 4;
  int hi = (kk >> 3) & 1;                       // lane half (K-half select)
  int e  = (kk & 7) + ((kk >= 16) ? 8 : 0);     // element within v16bf
  int lane = (n & 15) + hi * 16;
  WB[wb_frag(l, g, nt, kt) + (size_t)lane * 16 + e] = (__bf16)val;
}

// ---------------------------------------------------------------------------
// Kernel 2: hoisted input-side GEMM for one layer (software-pipelined).
// G[g][m][n] = X[m,:] @ W_i{g}[:,n] + b_i{g}[n], m = t*B+b over M_=32768.
// One wave = one 16(M) x 64(N) strip for one gate. bf16 out.
// ---------------------------------------------------------------------------
__global__ void __launch_bounds__(256) k_input_gemm(
    const float* X, const __bf16* WB,
    const float* b_ir, const float* b_iz, const float* b_in,
    __bf16* G, int layer) {
  int wave = threadIdx.x >> 5;
  int lane = threadIdx.x & 31;
  int W = blockIdx.x * 8 + wave;            // 0 .. 49151
  int mtile = W / 24;                       // 0 .. 2047
  int rem = W % 24;
  int g = rem >> 3;                         // 0..2  (ir, iz, in)
  int ng = rem & 7;                         // 0..7  (4 ntiles each)

  const float* bias = ((g == 0) ? b_ir : (g == 1) ? b_iz : b_in) + (size_t)layer * H_;

  v8f acc[4] = {{}, {}, {}, {}};
  int row = mtile * 16 + (lane & 15);
  int khalf = (lane >> 4) * 8;

  // software pipeline: fetch kt+1 while WMMAs for kt issue
  v16bf a = a_from_f32(X, row, khalf);
  v16bf bcur[4];
#pragma unroll
  for (int jj = 0; jj < 4; ++jj)
    bcur[jj] = b_load(WB, wb_frag(layer, g, ng * 4 + jj, 0), lane);

#pragma unroll 2
  for (int kt = 0; kt < 16; ++kt) {
    int ktn = (kt < 15) ? (kt + 1) : 15;     // clamped prefetch index
    v16bf an = a_from_f32(X, row, ktn * 32 + khalf);
    v16bf bnx[4];
#pragma unroll
    for (int jj = 0; jj < 4; ++jj)
      bnx[jj] = b_load(WB, wb_frag(layer, g, ng * 4 + jj, ktn), lane);
#pragma unroll
    for (int jj = 0; jj < 4; ++jj)
      acc[jj] = wmma_bf16(a, bcur[jj], acc[jj]);
    a = an;
#pragma unroll
    for (int jj = 0; jj < 4; ++jj) bcur[jj] = bnx[jj];
  }

  int col = lane & 15;
  int mrow0 = mtile * 16 + ((lane >> 4) * 8);
  size_t gbase = (size_t)g * M_ * H_;
#pragma unroll
  for (int jj = 0; jj < 4; ++jj) {
    int n = (ng * 4 + jj) * 16 + col;
    float bv = bias[n];
#pragma unroll
    for (int r = 0; r < 8; ++r) {
      G[gbase + (size_t)(mrow0 + r) * H_ + n] = (__bf16)(acc[jj][r] + bv);
    }
  }
}

// ---------------------------------------------------------------------------
// Kernel 3: persistent recurrent kernel for one layer.
// 16 blocks x 256 threads; block owns 32 N-columns. Per-block h-gate weights
// (96KB bf16) are DMA'd into LDS ONCE via TDM; each step the bf16 h_t image
// (64KB) is DMA'd into LDS via TDM. Inner loop is LDS-only -> WMMA.
// ---------------------------------------------------------------------------
__global__ void __launch_bounds__(256) k_recurrent(
    const __bf16* G, const __bf16* WB,
    const float* init_states,
    const float* b_hr, const float* b_hz, const float* b_hn,
    __bf16* hbf, float* out_hs, float* finals,
    unsigned* counter, int layer) {
  extern __shared__ char smem[];
  __bf16* shW = (__bf16*)smem;                    // [3][2][16][512] = 96KB
  __bf16* shH = (__bf16*)(smem + 3 * 2 * 16 * 512 * 2);  // [64][512] = 64KB

  int tid = threadIdx.x;
  int wave = tid >> 5, lane = tid & 31;
  const float* bhr = b_hr + (size_t)layer * H_;
  const float* bhz = b_hz + (size_t)layer * H_;
  const float* bhn = b_hn + (size_t)layer * H_;

  // init h(0) = bf16(init_states[layer]) into global double buffer slot 0
  for (int i = blockIdx.x * 256 + tid; i < BH_; i += NBLK_REC * 256)
    hbf[i] = (__bf16)init_states[(size_t)layer * BH_ + i];

  unsigned phase = 0;
  gbar(counter, ++phase);

  // one-time TDM: 3 gate-chunks of 16384 contiguous bf16 (2 ntiles x 16 kt x
  // 512), gate stride 262144 elems -> contiguous [g][ntl][kt][512] in LDS.
  unsigned shW_lds = (unsigned)(unsigned long long)(void*)shW;
  unsigned shH_lds = (unsigned)(unsigned long long)(void*)shH;
  if (wave == 0) {
    const __bf16* wsrc = WB + wb_frag(layer, 3, blockIdx.x * 2, 0);
    tdm_load(shW_lds, (unsigned long long)wsrc, 16384u, 3u,
             262144u, 3u, 262144ull);
  }

  int mtile = wave >> 1;                    // 0..3 (16 batch rows each)
  int ntl = wave & 1;                       // 2 ntiles per block
  int ntg = blockIdx.x * 2 + ntl;           // global ntile 0..31
  int col = lane & 15;
  int mrow0 = mtile * 16 + ((lane >> 4) * 8);
  int khalf = (lane >> 4) * 8;
  int n = ntg * 16 + col;
  float vbr = bhr[n], vbz = bhz[n], vbn = bhn[n];
  int arow = mtile * 16 + (lane & 15);

  for (int t = 0; t < T_; ++t) {
    const __bf16* cur = hbf + (size_t)(t & 1) * BH_;
    __bf16* nxt = hbf + (size_t)((t + 1) & 1) * BH_;

    // TDM: pull bf16 h_t image (64KB) into LDS; wave0 issues + waits,
    // everyone syncs. (First iteration's wait also covers the weight DMA.)
    if (wave == 0) {
      tdm_load(shH_lds, (unsigned long long)cur, 32768u, 0u,
               32768u, 1u, 32768ull);
      __builtin_amdgcn_s_wait_tensorcnt(0);
    }
    __syncthreads();

    // pull next step's precomputed input gates toward the caches
    if (t + 1 < T_) {
      size_t pgi = ((size_t)(t + 1) * B_ + mrow0) * H_ + n;
      __builtin_prefetch(G + pgi, 0, 1);
      __builtin_prefetch(G + (size_t)M_ * H_ + pgi, 0, 1);
      __builtin_prefetch(G + (size_t)2 * M_ * H_ + pgi, 0, 1);
    }

    v8f ar = {}, az = {}, an = {};
#pragma unroll 4
    for (int kt = 0; kt < 16; ++kt) {
      v16bf a  = a_from_bf16(shH, arow, kt * 32 + khalf);
      v16bf br = b_load(shW, (size_t)(((0 * 2 + ntl) * 16 + kt) * 512), lane);
      v16bf bz = b_load(shW, (size_t)(((1 * 2 + ntl) * 16 + kt) * 512), lane);
      v16bf bn = b_load(shW, (size_t)(((2 * 2 + ntl) * 16 + kt) * 512), lane);
      ar = wmma_bf16(a, br, ar);
      az = wmma_bf16(a, bz, az);
      an = wmma_bf16(a, bn, an);
    }

    // fused GRU gating on accumulator fragments
    size_t trow = (size_t)t * B_;
#pragma unroll
    for (int r = 0; r < 8; ++r) {
      int b = mrow0 + r;
      size_t gi = (trow + b) * H_ + n;
      float gir = (float)G[gi];
      float giz = (float)G[(size_t)M_ * H_ + gi];
      float gin = (float)G[(size_t)2 * M_ * H_ + gi];
      float hold = (float)shH[(size_t)b * H_ + n];
      float rv = sigmoidf_(gir + ar[r] + vbr);
      float zv = sigmoidf_(giz + az[r] + vbz);
      float nv = tanhf(gin + rv * (an[r] + vbn));
      float hnew = (1.0f - zv) * nv + zv * hold;
      nxt[(size_t)b * H_ + n] = (__bf16)hnew;
      out_hs[gi] = hnew;                    // [T,B,H] layout
      if (t == T_ - 1)
        finals[(size_t)layer * BH_ + (size_t)b * H_ + n] = hnew;
    }

    gbar(counter, ++phase);                 // publishes nxt; guards LDS reuse
  }
}

// ---------------------------------------------------------------------------
extern "C" void kernel_launch(void* const* d_in, const int* in_sizes, int n_in,
                              void* d_out, int out_size, void* d_ws, size_t ws_size,
                              hipStream_t stream) {
  const float* x    = (const float*)d_in[0];
  const float* init = (const float*)d_in[1];
  const float* w_ir = (const float*)d_in[2];
  const float* w_hr = (const float*)d_in[3];
  const float* w_iz = (const float*)d_in[4];
  const float* w_hz = (const float*)d_in[5];
  const float* w_in = (const float*)d_in[6];
  const float* w_hn = (const float*)d_in[7];
  const float* b_ir = (const float*)d_in[8];
  const float* b_hr = (const float*)d_in[9];
  const float* b_iz = (const float*)d_in[10];
  const float* b_hz = (const float*)d_in[11];
  const float* b_in = (const float*)d_in[12];
  const float* b_hn = (const float*)d_in[13];
  float* out = (float*)d_out;

  char* ws = (char*)d_ws;
  const size_t WB_BYTES  = (size_t)L_ * 6 * H_ * H_ * 2;   //   6 MB bf16 weights
  const size_t HS0_BYTES = (size_t)M_ * H_ * 4;            //  64 MB layer-0 outputs
  const size_t G_BYTES   = (size_t)3 * M_ * H_ * 2;        //  96 MB input gates
  const size_t HB_BYTES  = (size_t)2 * BH_ * 2;            // 128 KB bf16 h dbl buf

  __bf16*   WB   = (__bf16*)(void*)ws;
  float*    hs0  = (float*)(void*)(ws + WB_BYTES);
  __bf16*   G    = (__bf16*)(void*)(ws + WB_BYTES + HS0_BYTES);
  __bf16*   hbf  = (__bf16*)(void*)(ws + WB_BYTES + HS0_BYTES + G_BYTES);
  unsigned* ctr  = (unsigned*)(void*)(ws + WB_BYTES + HS0_BYTES + G_BYTES + HB_BYTES);

  float* finals = out + (size_t)M_ * H_;
  const size_t REC_LDS = (size_t)3 * 2 * 16 * 512 * 2 + (size_t)BH_ * 2; // 160KB

  // 1) weight swizzle (also resets grid-barrier counters)
  k_convert<<<(L_ * 6 * H_ * H_) / 256, 256, 0, stream>>>(
      w_ir, w_iz, w_in, w_hr, w_hz, w_hn, WB, ctr);

  // 2) layer 0: hoisted input GEMM, then persistent recurrence
  k_input_gemm<<<6144, 256, 0, stream>>>(x, WB, b_ir, b_iz, b_in, G, 0);
  k_recurrent<<<NBLK_REC, 256, REC_LDS, stream>>>(G, WB, init, b_hr, b_hz, b_hn,
                                                  hbf, hs0, finals, ctr + 0, 0);

  // 3) layer 1: same, consuming layer-0 outputs; writes d_out directly
  k_input_gemm<<<6144, 256, 0, stream>>>(hs0, WB, b_ir, b_iz, b_in, G, 1);
  k_recurrent<<<NBLK_REC, 256, REC_LDS, stream>>>(G, WB, init, b_hr, b_hz, b_hn,
                                                  hbf, out, finals, ctr + 16, 1);
}
